// FBSNN_44298292691244
// MI455X (gfx1250) — matrix operand order repaired and checked by reading, not compiled, not run on hardware
//
#include <hip/hip_runtime.h>
#include <hip/hip_bf16.h>

// ---------------- problem constants ----------------
#define M_PATHS 2048
#define NSTEP   50
#define NT      51            // N_STEPS + 1
#define DDIM    100
#define HID     256
#define KIN     128           // input features (1+100) zero-padded to 128
#define BROWS   32            // rows per workgroup (2 waves x 16 rows)
#define NROWS   (M_PATHS * NT)   // 104448 total MLP rows
#define SIGMA   0.4f
#define RRATE   0.05f
#define DT      (1.0f / 50.0f)

// d_out layout: [loss | X_stack | Y_stack | Z_stack], fp32
#define OX 1
#define NX (M_PATHS * NT * DDIM)
#define OY (OX + NX)
#define OZ (OY + M_PATHS * NT)

// padded LDS row strides (halves) to spread LDS banks
#define LDH 264
#define LDI 136

// f16 weight scratch layout (element offsets into d_ws)
#define W1T_OFF 0        // [256][128]
#define W1F_OFF 32768    // [128][256]
#define W2T_OFF 65536    // [256][256]
#define W2F_OFF 131072
#define W3T_OFF 196608
#define W3F_OFF 262144
#define W4T_OFF 327680
#define W4F_OFF 393216

typedef __attribute__((ext_vector_type(16))) _Float16 v16h;
typedef __attribute__((ext_vector_type(8)))  float    v8f;

// ---------------- WMMA fragment helpers ----------------
// A-fragment (16x32 f16, MxK) from LDS, row-major with stride ldh.
// Layout per ISA 7.12.2: lanes 0-15 -> M=lane, K groups 0..7/16..23;
// lanes 16-31 -> M=lane-16, K groups 8..15/24..31; element i -> VGPR i/2 half i%2.
__device__ __forceinline__ v16h lds_frag_a(const _Float16* h, int ldh, int mr, int kb, int lane) {
  const int Mrow = mr + (lane & 15);
  const int koff = (lane & 16) ? 8 : 0;
  const _Float16* row = h + Mrow * ldh + kb;
  v16h a;
#pragma unroll
  for (int i = 0; i < 16; ++i) {
    const int j = i >> 1, p = i & 1;
    const int k = (j < 4) ? (koff + 2 * j + p) : (16 + koff + 2 * (j - 4) + p);
    a[i] = row[k];
  }
  return a;
}

// B-fragment (32x16 f16, KxN): lanes 0-15 hold K=0..7/16..23, lanes 16-31 K=8..15/24..31,
// element i fast along K. Weight stored so K-run is contiguous: w[(col)*ld + row].
__device__ __forceinline__ v16h glb_frag_b(const _Float16* w, int ld, int colb, int rowb, int lane) {
  const _Float16* p = w + (colb + (lane & 15)) * ld + rowb + ((lane & 16) ? 8 : 0);
  v16h b;
#pragma unroll
  for (int i = 0; i < 16; ++i) b[i] = p[(i < 8) ? i : (i + 8)];
  return b;
}

__device__ __forceinline__ v8f wmma_f16(v16h a, v16h b, v8f c) {
  // 8 args: (neg_a, A, neg_b, B, c_mod, C, reuse_a, reuse_b)
  return __builtin_amdgcn_wmma_f32_16x16x32_f16(false, a, false, b, (short)0, c, false, false);
}

// ---------------- fused MLP layers ----------------
// forward: z = h @ W + b ; hout = sin(z) ; zc = cos(z)   (wave-private rows)
template<int KDIM, int LDIN>
__device__ __forceinline__ void fwd_layer(const _Float16* hin, _Float16* hout, _Float16* zc,
                                          const _Float16* Wt, const float* bias,
                                          int mr, int lane) {
#pragma unroll 1
  for (int nb = 0; nb < HID; nb += 16) {
    v8f acc = {};
#pragma unroll
    for (int kb = 0; kb < KDIM; kb += 32) {
      v16h a = lds_frag_a(hin, LDIN, mr, kb, lane);
      v16h b = glb_frag_b(Wt, KDIM, nb, kb, lane);
      acc = wmma_f16(a, b, acc);
    }
    const int N  = nb + (lane & 15);
    const int Mb = mr + ((lane & 16) ? 8 : 0);
    const float bz = bias[N];
#pragma unroll
    for (int r = 0; r < 8; ++r) {
      const float z = acc[r] + bz;
      hout[(Mb + r) * LDH + N] = (_Float16)__sinf(z);
      zc  [(Mb + r) * LDH + N] = (_Float16)__cosf(z);
    }
  }
}

// backward: dz_prev = (dz_cur @ W^T) * cos(z_prev)
__device__ __forceinline__ void bwd_layer(const _Float16* dzin, _Float16* dzout,
                                          const _Float16* czprev, const _Float16* Wf,
                                          int mr, int lane) {
#pragma unroll 1
  for (int nb = 0; nb < HID; nb += 16) {
    v8f acc = {};
#pragma unroll
    for (int kb = 0; kb < HID; kb += 32) {
      v16h a = lds_frag_a(dzin, LDH, mr, kb, lane);
      v16h b = glb_frag_b(Wf, HID, nb, kb, lane);   // Wf[kprev*256 + n] == W[kprev][n]
      acc = wmma_f16(a, b, acc);
    }
    const int N  = nb + (lane & 15);
    const int Mb = mr + ((lane & 16) ? 8 : 0);
#pragma unroll
    for (int r = 0; r < 8; ++r) {
      const int off = (Mb + r) * LDH + N;
      dzout[off] = (_Float16)(acc[r] * (float)czprev[off]);
    }
  }
}

// ---------------- kernels ----------------
// X path: X_{n+1} = X_n * (1 + sigma * dW_n), dW from Brownian path W.
__global__ void fbsnn_path_kernel(const float* __restrict__ Wpath, const float* __restrict__ Xi,
                                  float* __restrict__ out) {
  const int tid = blockIdx.x * blockDim.x + threadIdx.x;
  if (tid >= M_PATHS * DDIM) return;
  const int m = tid / DDIM, d = tid % DDIM;
  float x = Xi[m * DDIM + d];
  out[OX + (m * NT) * DDIM + d] = x;
  const float* wr = Wpath + (m * NT) * DDIM + d;
#pragma unroll 1
  for (int n = 0; n < NSTEP; ++n) {
    const float dw = wr[(n + 1) * DDIM] - wr[n * DDIM];
    x = x * (1.f + SIGMA * dw);
    out[OX + (m * NT + n + 1) * DDIM + d] = x;
  }
}

// weights -> f16 in both orientations (Wt for forward B-frags, Wf for backward)
__global__ void fbsnn_prep_kernel(const float* __restrict__ W1, const float* __restrict__ W2,
                                  const float* __restrict__ W3, const float* __restrict__ W4,
                                  _Float16* __restrict__ wsh) {
  const int k = blockIdx.x;   // 0..255 (fan-in index)
  const int n = threadIdx.x;  // 0..255 (fan-out index)
  if (k < KIN) {
    const float v = (k < 101) ? W1[k * HID + n] : 0.f;
    wsh[W1T_OFF + n * KIN + k] = (_Float16)v;
    wsh[W1F_OFF + k * HID + n] = (_Float16)v;
  }
  const float v2 = W2[k * HID + n];
  wsh[W2T_OFF + n * HID + k] = (_Float16)v2;
  wsh[W2F_OFF + k * HID + n] = (_Float16)v2;
  const float v3 = W3[k * HID + n];
  wsh[W3T_OFF + n * HID + k] = (_Float16)v3;
  wsh[W3F_OFF + k * HID + n] = (_Float16)v3;
  const float v4 = W4[k * HID + n];
  wsh[W4T_OFF + n * HID + k] = (_Float16)v4;
  wsh[W4F_OFF + k * HID + n] = (_Float16)v4;
}

// fused forward + VJP over 32 rows per block; each wave privately owns 16 rows
// (no barriers needed), activations + cos(z) never leave LDS.
__launch_bounds__(64)
__global__ void fbsnn_mlp_kernel(const _Float16* __restrict__ wsh,
                                 const float* __restrict__ b1, const float* __restrict__ b2,
                                 const float* __restrict__ b3, const float* __restrict__ b4,
                                 const float* __restrict__ W5, const float* __restrict__ b5,
                                 float* __restrict__ out) {
  __shared__ _Float16 hA [BROWS * LDH];
  __shared__ _Float16 hB [BROWS * LDH];
  __shared__ _Float16 cz0[BROWS * LDH];
  __shared__ _Float16 cz1[BROWS * LDH];
  __shared__ _Float16 cz2[BROWS * LDH];
  __shared__ _Float16 cz3[BROWS * LDH];

  const _Float16* W1t = wsh + W1T_OFF;
  const _Float16* W1f = wsh + W1F_OFF;
  const _Float16* W2t = wsh + W2T_OFF;
  const _Float16* W2f = wsh + W2F_OFF;
  const _Float16* W3t = wsh + W3T_OFF;
  const _Float16* W3f = wsh + W3F_OFF;
  const _Float16* W4t = wsh + W4T_OFF;
  const _Float16* W4f = wsh + W4F_OFF;

  const int rb   = blockIdx.x * BROWS;
  const int lane = threadIdx.x & 31;
  const int mr   = (threadIdx.x >> 5) * 16;

  // stage padded input rows [t_n, X(m,n,:), 0...] for this wave's 16 rows
  for (int idx = lane; idx < 16 * KIN; idx += 32) {
    const int lr = mr + (idx >> 7);
    const int k  = idx & (KIN - 1);
    const int b  = rb + lr;
    float v = 0.f;
    if (k == 0)            v = (float)(b % NT) * DT;
    else if (k <= DDIM)    v = out[OX + b * DDIM + (k - 1)];
    hA[lr * LDI + k] = (_Float16)v;
  }

  // forward (ping-pong hA/hB), keep cos(z_l) for the VJP
  fwd_layer<KIN, LDI>(hA, hB, cz0, W1t, b1, mr, lane);
  fwd_layer<HID, LDH>(hB, hA, cz1, W2t, b2, mr, lane);
  fwd_layer<HID, LDH>(hA, hB, cz2, W3t, b3, mr, lane);
  fwd_layer<HID, LDH>(hB, hA, cz3, W4t, b4, mr, lane);   // h4 -> hA

  // head: y = h4 @ W5 + b5
  if (lane < 16) {
    const int lr = mr + lane;
    float y = b5[0];
#pragma unroll 4
    for (int k = 0; k < HID; ++k) y += (float)hA[lr * LDH + k] * W5[k];
    out[OY + rb + lr] = y;
  }

  // VJP seed: dz4 = cos(z4) * W5 (row broadcast of W5)
  for (int idx = lane; idx < 16 * HID; idx += 32) {
    const int lr = mr + (idx >> 8);
    const int k  = idx & (HID - 1);
    hB[lr * LDH + k] = (_Float16)((float)cz3[lr * LDH + k] * W5[k]);
  }

  bwd_layer(hB, hA, cz2, W4f, mr, lane);   // dz3
  bwd_layer(hA, hB, cz1, W3f, mr, lane);   // dz2
  bwd_layer(hB, hA, cz0, W2f, mr, lane);   // dz1 -> hA

  // gin = dz1 @ W1^T; Du = gin[:,1..100]; fuse Z = sigma * X * Du
#pragma unroll 1
  for (int nb = 0; nb < KIN; nb += 16) {
    v8f acc = {};
#pragma unroll
    for (int kb = 0; kb < HID; kb += 32) {
      v16h a = lds_frag_a(hA, LDH, mr, kb, lane);
      v16h b = glb_frag_b(W1f, HID, nb, kb, lane);
      acc = wmma_f16(a, b, acc);
    }
    const int c  = nb + (lane & 15);
    const int Mb = mr + ((lane & 16) ? 8 : 0);
    if (c >= 1 && c <= DDIM) {
      const int d = c - 1;
#pragma unroll
      for (int r = 0; r < 8; ++r) {
        const int b = rb + Mb + r;
        const float x = out[OX + b * DDIM + d];
        out[OZ + b * DDIM + d] = SIGMA * x * acc[r];
      }
    }
  }
}

// deterministic loss: all scan residuals are independent once X/Y/Z exist.
__global__ void fbsnn_loss_kernel(const float* __restrict__ Wpath, float* __restrict__ out) {
  __shared__ float red[256];
  const int tid = threadIdx.x;
  float acc = 0.f;
  for (int m = tid; m < M_PATHS; m += 256) {
    const float* Yr = out + OY + m * NT;
#pragma unroll 1
    for (int n = 0; n < NSTEP; ++n) {
      const float* Xr = out + OX + (m * NT + n) * DDIM;
      const float* Zr = out + OZ + (m * NT + n) * DDIM;
      const float* W0 = Wpath + (m * NT + n) * DDIM;
      float zdw = 0.f, xz = 0.f;
      for (int d = 0; d < DDIM; ++d) {
        zdw += Zr[d] * (W0[d + DDIM] - W0[d]);
        xz  += Xr[d] * Zr[d];
      }
      const float phi = RRATE * (Yr[n] - xz);
      const float yt  = Yr[n] + phi * DT + zdw;
      const float r   = Yr[n + 1] - yt;
      acc += r * r;
    }
    const float* X1 = out + OX + (m * NT + NSTEP) * DDIM;
    const float* Z1 = out + OZ + (m * NT + NSTEP) * DDIM;
    float g = 0.f;
    for (int d = 0; d < DDIM; ++d) g += X1[d] * X1[d];
    const float r1 = Yr[NSTEP] - g;
    acc += r1 * r1;
    for (int d = 0; d < DDIM; ++d) {
      const float zt = 2.f * SIGMA * X1[d] * X1[d];
      const float rr = Z1[d] - zt;
      acc += rr * rr;
    }
  }
  red[tid] = acc;
  __syncthreads();
  for (int s = 128; s > 0; s >>= 1) {
    if (tid < s) red[tid] += red[tid + s];
    __syncthreads();
  }
  if (tid == 0) out[0] = red[0] / (float)NSTEP;
}

// ---------------- launch ----------------
extern "C" void kernel_launch(void* const* d_in, const int* in_sizes, int n_in,
                              void* d_out, int out_size, void* d_ws, size_t ws_size,
                              hipStream_t stream) {
  (void)in_sizes; (void)n_in; (void)out_size; (void)ws_size;
  // inputs: t, W, Xi, W1, b1, W2, b2, W3, b3, W4, b4, W5, b5
  const float* Wpath = (const float*)d_in[1];
  const float* Xi    = (const float*)d_in[2];
  const float* W1 = (const float*)d_in[3];  const float* b1 = (const float*)d_in[4];
  const float* W2 = (const float*)d_in[5];  const float* b2 = (const float*)d_in[6];
  const float* W3 = (const float*)d_in[7];  const float* b3 = (const float*)d_in[8];
  const float* W4 = (const float*)d_in[9];  const float* b4 = (const float*)d_in[10];
  const float* W5 = (const float*)d_in[11]; const float* b5 = (const float*)d_in[12];
  float* out = (float*)d_out;
  _Float16* wsh = (_Float16*)d_ws;   // ~900 KB of f16 weight copies

  fbsnn_prep_kernel<<<256, 256, 0, stream>>>(W1, W2, W3, W4, wsh);
  fbsnn_path_kernel<<<(M_PATHS * DDIM + 255) / 256, 256, 0, stream>>>(Wpath, Xi, out);
  fbsnn_mlp_kernel<<<NROWS / BROWS, 64, 0, stream>>>(wsh, b1, b2, b3, b4, W5, b5, out);
  fbsnn_loss_kernel<<<1, 256, 0, stream>>>(Wpath, out);
}
